// LSTM_33363305955791
// MI455X (gfx1250) — matrix operand order repaired and checked by reading, not compiled, not run on hardware
//
#include <hip/hip_runtime.h>

typedef __attribute__((ext_vector_type(16))) _Float16 v16h;
typedef __attribute__((ext_vector_type(8)))  _Float16 v8h;
typedef __attribute__((ext_vector_type(8)))  float    v8f;

#define B_  128
#define N_  2
#define T_  256
#define H_  128
#define G4  512   // 4*H

// ---------------- workspace layout (bytes) ----------------
static constexpr size_t OFF_WHH0 = 0;                                   // 4*(512*128)*2  = 512 KB
static constexpr size_t OFF_WHH1 = OFF_WHH0 + 4ull * G4 * H_ * 2;       // 512 KB
static constexpr size_t OFF_WIH1 = OFF_WHH1 + 4ull * G4 * H_ * 2;       // 4*(512*256)*2 = 1 MB
static constexpr size_t OFF_Y0   = OFF_WIH1 + 4ull * G4 * 256 * 2;      // [N][T][B][256] f16 = 32 MB
static constexpr size_t OFF_XP1  = OFF_Y0   + (size_t)N_ * T_ * B_ * 256 * 2; // [4][T][65536] f16 = 128 MB
static constexpr size_t OFF_HL   = OFF_XP1  + 4ull * T_ * 65536 * 2;    // [4][B][H] f32 = 256 KB

// LDS partition for the persistent LSTM kernels (dynamic shared memory:
// 164 KB < 320 KB/WGP; dynamic so we bypass the 64 KB static-LDS check).
#define WLDS_BYTES (G4 * H_ * 2)            // 131072
#define HLDS_BYTES (B_ * H_ * 2)            // 32768
#define LSTM_SMEM  (WLDS_BYTES + HLDS_BYTES + B_ * 4)

__device__ __forceinline__ float sig_(float x) {
  x = fminf(fmaxf(x, -30.f), 30.f);
  return 1.0f / (1.0f + __expf(-x));
}
__device__ __forceinline__ float tanh_(float x) {
  x = fminf(fmaxf(x, -15.f), 15.f);
  float e = __expf(2.0f * x);
  return (e - 1.0f) / (e + 1.0f);
}

// Plain (non-volatile) 32-byte LDS load of one WMMA B operand through an
// explicit addrspace(3) pointer -> ds_load_b128 pair with normal, batchable
// s_wait_dscnt handling. Hoisting is prevented at the call sites by passing
// an asm-opaqued base offset, NOT by volatile (volatile previously lowered
// to serialized flat_load_b128 scope:SCOPE_SYS + s_wait_loadcnt 0 each).
typedef __attribute__((address_space(3))) const v8h* lds_v8h_cp;
__device__ __forceinline__ v16h load_b_tile(unsigned byteOff) {
  lds_v8h_cp p = (lds_v8h_cp)byteOff;
  v8h lo = p[0];
  v8h hi = p[1];
  return __builtin_shufflevector(lo, hi, 0, 1, 2, 3, 4, 5, 6, 7, 8, 9, 10, 11,
                                 12, 13, 14, 15);
}

// Launder a 32-bit LDS offset through an empty volatile asm: the value is
// unchanged but LICM/CSE cannot prove it, so weight loads stay inside the
// recurrence loop instead of ballooning into a 512-VGPR live range.
__device__ __forceinline__ unsigned opaque_u32(unsigned v) {
  asm volatile("" : "+v"(v));
  return v;
}

// Async DMA of the swizzled weight matrix (128 KB) into LDS.
// Generic shared pointers carry the LDS byte offset in their low 32 bits
// so we can hand the offset VGPR to
// global_load_async_to_lds_b128 directly. Tracked by ASYNCcnt.
__device__ __forceinline__ void async_copy_weights(_Float16* wlds,
                                                   const _Float16* wB,
                                                   int tid) {
  unsigned lbase = (unsigned)(uintptr_t)(void*)wlds;
  for (int i = tid; i < (G4 * H_) / 8; i += 512) {
    unsigned loff = lbase + (unsigned)i * 16u;
    unsigned long long gp = (unsigned long long)(uintptr_t)(wB + (size_t)i * 8);
    asm volatile("global_load_async_to_lds_b128 %0, %1, off"
                 :
                 : "v"(loff), "v"(gp)
                 : "memory");
  }
  asm volatile("s_wait_asynccnt 0" ::: "memory");
}

// --------------------------------------------------------------------------
// Repack a [512][K] fp32 gate-weight matrix (gates = act @ W^T) into f16
// WMMA B-matrix tiles: tile(kc,nt) is 1024 bytes, lane L owns 16 halves at
// tile*512 + L*16; element j of lane L is Bmat[k = kc*32 + (L/16)*16 + j]
// [n = nt*16 + L%16]  (contiguous k order, same as the A-tile loads).
// --------------------------------------------------------------------------
__global__ void swizzleB_kernel(const float* __restrict__ W,
                                _Float16* __restrict__ dst,
                                int K, int nSlices) {
  int total = nSlices * G4 * K;
  for (int idx = blockIdx.x * blockDim.x + threadIdx.x; idx < total;
       idx += gridDim.x * blockDim.x) {
    int slice = idx / (G4 * K);
    int e     = idx - slice * (G4 * K);
    int j     = e & 15;
    int L     = (e >> 4) & 31;
    int tile  = e >> 9;           // 512 halves per tile
    int nt    = tile & 31;        // 32 n-tiles (512/16)
    int kc    = tile >> 5;        // K/32 k-chunks
    int k     = kc * 32 + (L >> 4) * 16 + j;
    int n     = nt * 16 + (L & 15);
    dst[(size_t)slice * (G4 * K) + (size_t)tile * 512 + L * 16 + j] =
        (_Float16)W[(size_t)slice * (G4 * K) + (size_t)n * K + k];
  }
}

// --------------------------------------------------------------------------
// Layer-0 persistent bidirectional LSTM. One workgroup per (node,dir),
// 16 waves. Wave w owns m-tile = w/2 (16 batch rows) and h-tiles
// hbase..hbase+3. c-state lives in v8f registers for the whole sequence;
// h-state round-trips through LDS in f16; weights are LDS-resident.
// --------------------------------------------------------------------------
__global__ void __launch_bounds__(512)
lstm0_kernel(const float* __restrict__ x,     // [B][N][T]
             const float* __restrict__ h0,    // [N][4][B][H]
             const float* __restrict__ c0,    // [N][4][B][H]
             const float* __restrict__ wih0,  // [N][2][512][1]
             const float* __restrict__ b0,    // [N][2][512]
             const _Float16* __restrict__ whh0s,  // swizzled
             _Float16* __restrict__ y0)       // [N][T][B][256]
{
  extern __shared__ char smem[];
  _Float16* wlds = (_Float16*)smem;
  _Float16* hlds = (_Float16*)(smem + WLDS_BYTES);
  float*    xcol = (float*)(smem + WLDS_BYTES + HLDS_BYTES);

  const int nd   = blockIdx.x;
  const int node = nd >> 1, dir = nd & 1;
  const int tid  = threadIdx.x;
  const int lane = tid & 31, wid = tid >> 5;
  const int m    = wid >> 1;
  const int hbase = (wid & 1) * 4;
  const int hi   = lane >> 4, nl = lane & 15;

  async_copy_weights(wlds, whh0s + (size_t)nd * (G4 * H_), tid);

  for (int i = tid; i < B_ * H_; i += 512) {
    int r = i >> 7, c = i & 127;
    hlds[r * H_ + c] = (_Float16)h0[(((size_t)node * 4 + dir) * B_ + r) * H_ + c];
  }

  v8f  creg[4];
  float wv[4][4], bvv[4][4];
#pragma unroll
  for (int bi = 0; bi < 4; ++bi) {
    int hn = hbase + bi;
#pragma unroll
    for (int r = 0; r < 8; ++r) {
      int row = m * 16 + r + 8 * hi;
      creg[bi][r] = c0[(((size_t)node * 4 + dir) * B_ + row) * H_ + hn * 16 + nl];
    }
#pragma unroll
    for (int g = 0; g < 4; ++g) {
      int n = g * 128 + hn * 16 + nl;
      wv[bi][g]  = wih0[(size_t)nd * G4 + n];
      bvv[bi][g] = b0[(size_t)nd * G4 + n];
    }
  }
  __syncthreads();

  // base LDS byte offset of this lane's B-operand slice (lane*32 inside tile)
  const unsigned wbase =
      (unsigned)(uintptr_t)(void*)wlds + (unsigned)lane * 32u;
  const v8f vzero = {0.f, 0.f, 0.f, 0.f, 0.f, 0.f, 0.f, 0.f};

#pragma unroll 1
  for (int s = 0; s < T_; ++s) {
    int t = dir ? (T_ - 1 - s) : s;
    if (tid < B_) xcol[tid] = x[((size_t)tid * N_ + node) * T_ + t];

    // A tiles (h state), contiguous k order: M = lane%16, K split by half-wave
    v16h a[4];
    {
      const _Float16* hp = hlds + (m * 16 + nl) * H_;
#pragma unroll
      for (int kc = 0; kc < 4; ++kc) {
        int cbase = kc * 32 + hi * 16;
        v8h lo = *(const v8h*)(hp + cbase);
        v8h hi8 = *(const v8h*)(hp + cbase + 8);
        a[kc] = __builtin_shufflevector(lo, hi8, 0, 1, 2, 3, 4, 5, 6, 7, 8, 9,
                                        10, 11, 12, 13, 14, 15);
      }
    }
    __syncthreads();   // everyone has read old h / sees xcol

#pragma unroll
    for (int bi = 0; bi < 4; ++bi) {
      int hn = hbase + bi;
      unsigned wo = opaque_u32(wbase);   // fresh per block: no hoist/CSE
      v8f acc[4];
#pragma unroll
      for (int g = 0; g < 4; ++g) acc[g] = vzero;
#pragma unroll
      for (int g = 0; g < 4; ++g)
#pragma unroll
        for (int kc = 0; kc < 4; ++kc) {
          v16h b =
              load_b_tile(wo + (unsigned)(kc * 32 + g * 8 + hn) * 1024u);
          acc[g] = __builtin_amdgcn_wmma_f32_16x16x32_f16(
              false, a[kc], false, b, (short)0, acc[g], false, false);
        }
#pragma unroll
      for (int r = 0; r < 8; ++r) {
        int row = m * 16 + r + 8 * hi;
        float xv = xcol[row];
        float gi = acc[0][r] + xv * wv[bi][0] + bvv[bi][0];
        float gf = acc[1][r] + xv * wv[bi][1] + bvv[bi][1];
        float gg = acc[2][r] + xv * wv[bi][2] + bvv[bi][2];
        float go = acc[3][r] + xv * wv[bi][3] + bvv[bi][3];
        float cc = sig_(gf) * creg[bi][r] + sig_(gi) * tanh_(gg);
        creg[bi][r] = cc;
        float hh = sig_(go) * tanh_(cc);
        hlds[row * H_ + hn * 16 + nl] = (_Float16)hh;
        y0[(((size_t)node * T_ + t) * B_ + row) * 256 + dir * 128 + hn * 16 +
           nl] = (_Float16)hh;
      }
    }
    __syncthreads();   // new h visible before next A loads
  }
}

// --------------------------------------------------------------------------
// xp1 = y0 @ w_ih1^T + b1, per (node,dir,t): [128x256]·[256x512].
// Output stored in C/D-tile swizzled f16 (lane L owns its 8 accumulator
// elements contiguously) so lstm1 can C-init with one 16-byte load.
// --------------------------------------------------------------------------
__global__ void __launch_bounds__(512)
xp1_kernel(const _Float16* __restrict__ y0,
           const _Float16* __restrict__ wih1s,
           const float* __restrict__ b1,
           _Float16* __restrict__ xp1s)
{
  const int bid = blockIdx.x;
  const int nd = bid / T_, t = bid % T_;
  const int node = nd >> 1;
  const int tid = threadIdx.x, lane = tid & 31, wid = tid >> 5;
  const int m = wid >> 1, ntb = (wid & 1) * 16;
  const int hi = lane >> 4, nl = lane & 15;

  const _Float16* yrow =
      y0 + (((size_t)node * T_ + t) * B_ + m * 16 + nl) * 256;
  v16h a[8];
#pragma unroll
  for (int kc = 0; kc < 8; ++kc) {
    int cbase = kc * 32 + hi * 16;
    v8h lo = *(const v8h*)(yrow + cbase);
    v8h hi8 = *(const v8h*)(yrow + cbase + 8);
    a[kc] = __builtin_shufflevector(lo, hi8, 0, 1, 2, 3, 4, 5, 6, 7, 8, 9, 10,
                                    11, 12, 13, 14, 15);
  }

  const _Float16* wB = wih1s + (size_t)nd * (G4 * 256);
  _Float16* outp = xp1s + ((size_t)nd * T_ + t) * 65536;

#pragma unroll 1
  for (int q = 0; q < 16; ++q) {
    int nt = ntb + q;
    float bias = b1[(size_t)nd * G4 + nt * 16 + nl];
    v8f acc = {bias, bias, bias, bias, bias, bias, bias, bias};
#pragma unroll
    for (int kc = 0; kc < 8; ++kc) {
      v16h b = *(const v16h*)(wB + ((size_t)(kc * 32 + nt)) * 512 + lane * 16);
      acc = __builtin_amdgcn_wmma_f32_16x16x32_f16(false, a[kc], false, b,
                                                   (short)0, acc, false, false);
    }
    v8h o;
#pragma unroll
    for (int r = 0; r < 8; ++r) o[r] = (_Float16)acc[r];
    *(v8h*)(outp + ((size_t)(m * 32 + nt)) * 256 + lane * 8) = o;
  }
}

// --------------------------------------------------------------------------
// Layer-1 persistent LSTM. Gate accumulators are C-initialized from xp1
// tiles. Forward runs T steps; backward needs only 1 step, because the head
// consumes only y[:, -1] and hb[::-1][T-1] == backward step 0.
// --------------------------------------------------------------------------
__global__ void __launch_bounds__(512)
lstm1_kernel(const float* __restrict__ h0,
             const float* __restrict__ c0,
             const _Float16* __restrict__ whh1s,
             const _Float16* __restrict__ xp1s,
             float* __restrict__ hlast)     // [4][B][H]
{
  extern __shared__ char smem[];
  _Float16* wlds = (_Float16*)smem;
  _Float16* hlds = (_Float16*)(smem + WLDS_BYTES);

  const int nd   = blockIdx.x;
  const int node = nd >> 1, dir = nd & 1;
  const int tid  = threadIdx.x;
  const int lane = tid & 31, wid = tid >> 5;
  const int m    = wid >> 1;
  const int hbase = (wid & 1) * 4;
  const int hi   = lane >> 4, nl = lane & 15;

  async_copy_weights(wlds, whh1s + (size_t)nd * (G4 * H_), tid);

  for (int i = tid; i < B_ * H_; i += 512) {
    int r = i >> 7, c = i & 127;
    hlds[r * H_ + c] =
        (_Float16)h0[(((size_t)node * 4 + 2 + dir) * B_ + r) * H_ + c];
  }

  v8f creg[4];
#pragma unroll
  for (int bi = 0; bi < 4; ++bi) {
    int hn = hbase + bi;
#pragma unroll
    for (int r = 0; r < 8; ++r) {
      int row = m * 16 + r + 8 * hi;
      creg[bi][r] =
          c0[(((size_t)node * 4 + 2 + dir) * B_ + row) * H_ + hn * 16 + nl];
    }
  }
  __syncthreads();

  const unsigned wbase =
      (unsigned)(uintptr_t)(void*)wlds + (unsigned)lane * 32u;
  const int steps = dir ? 1 : T_;

#pragma unroll 1
  for (int s = 0; s < steps; ++s) {
    int t = dir ? (T_ - 1 - s) : s;
    const _Float16* xpt = xp1s + ((size_t)nd * T_ + t) * 65536;
    if (s + 1 < steps)  // stream next timestep's projections into L2
      __builtin_prefetch(xp1s + ((size_t)nd * T_ + (t + 1)) * 65536 +
                             (size_t)tid * 128, 0, 1);

    v16h a[4];
    {
      const _Float16* hp = hlds + (m * 16 + nl) * H_;
#pragma unroll
      for (int kc = 0; kc < 4; ++kc) {
        int cbase = kc * 32 + hi * 16;
        v8h lo = *(const v8h*)(hp + cbase);
        v8h hi8 = *(const v8h*)(hp + cbase + 8);
        a[kc] = __builtin_shufflevector(lo, hi8, 0, 1, 2, 3, 4, 5, 6, 7, 8, 9,
                                        10, 11, 12, 13, 14, 15);
      }
    }
    __syncthreads();

#pragma unroll
    for (int bi = 0; bi < 4; ++bi) {
      int hn = hbase + bi;
      unsigned wo = opaque_u32(wbase);   // fresh per block: no hoist/CSE
      v8f acc[4];
#pragma unroll
      for (int g = 0; g < 4; ++g) {
        v8h xv = *(const v8h*)(xpt + ((size_t)(m * 32 + g * 8 + hn)) * 256 +
                               lane * 8);
#pragma unroll
        for (int r = 0; r < 8; ++r) acc[g][r] = (float)xv[r];
      }
#pragma unroll
      for (int g = 0; g < 4; ++g)
#pragma unroll
        for (int kc = 0; kc < 4; ++kc) {
          v16h b =
              load_b_tile(wo + (unsigned)(kc * 32 + g * 8 + hn) * 1024u);
          acc[g] = __builtin_amdgcn_wmma_f32_16x16x32_f16(
              false, a[kc], false, b, (short)0, acc[g], false, false);
        }
#pragma unroll
      for (int r = 0; r < 8; ++r) {
        int row = m * 16 + r + 8 * hi;
        float cc = sig_(acc[1][r]) * creg[bi][r] +
                   sig_(acc[0][r]) * tanh_(acc[2][r]);
        creg[bi][r] = cc;
        float hh = sig_(acc[3][r]) * tanh_(cc);
        hlds[row * H_ + hn * 16 + nl] = (_Float16)hh;
      }
    }
    __syncthreads();
  }

  for (int i = tid; i < B_ * H_; i += 512) {
    int r = i >> 7, c = i & 127;
    hlast[((size_t)nd * B_ + r) * H_ + c] = (float)hlds[r * H_ + c];
  }
}

// --------------------------------------------------------------------------
// Fused MLP head: fc0 -> residual LayerNorm(ddof=1) ReLU -> fc1 -> same ->
// fc2. One block per batch row. Tiny FLOPs; VALU path is fine.
// --------------------------------------------------------------------------
__device__ __forceinline__ float block_sum256(float v, float* red, int j) {
  red[j] = v;
  __syncthreads();
  for (int off = 128; off > 0; off >>= 1) {
    if (j < off) red[j] += red[j + off];
    __syncthreads();
  }
  float s = red[0];
  __syncthreads();
  return s;
}

__global__ void __launch_bounds__(256)
head_kernel(const float* __restrict__ hlast,
            const float* __restrict__ w0, const float* __restrict__ bb0,
            const float* __restrict__ a0, const float* __restrict__ g0,
            const float* __restrict__ w1, const float* __restrict__ bb1,
            const float* __restrict__ a1, const float* __restrict__ g1,
            const float* __restrict__ w2, const float* __restrict__ bb2,
            float* __restrict__ out)
{
  __shared__ float rowv[512];
  __shared__ float tv[256];
  __shared__ float red[256];
  const int b = blockIdx.x, j = threadIdx.x;

  for (int c = j; c < 512; c += 256) {
    int node = c >> 8, rem = c & 255, d = rem >> 7, h = rem & 127;
    rowv[c] = hlast[(((size_t)node * 2 + d) * B_ + b) * H_ + h];
  }
  __syncthreads();

  float u = bb0[j];
  for (int k = 0; k < 512; ++k) u += rowv[k] * w0[k * 256 + j];
  {
    float s1 = block_sum256(u, red, j);
    float s2 = block_sum256(u * u, red, j);
    float mu = s1 * (1.0f / 256.0f);
    float var = (s2 - 256.0f * mu * mu) * (1.0f / 255.0f);
    float sd = sqrtf(fmaxf(var, 0.0f)) + 1e-6f;
    float ln = (u - mu) / sd * a0[j] + g0[j];
    tv[j] = fmaxf(u + ln, 0.0f);
  }
  __syncthreads();

  float u1 = bb1[j];
  for (int k = 0; k < 256; ++k) u1 += tv[k] * w1[k * 256 + j];
  {
    float s1 = block_sum256(u1, red, j);
    float s2 = block_sum256(u1 * u1, red, j);
    float mu = s1 * (1.0f / 256.0f);
    float var = (s2 - 256.0f * mu * mu) * (1.0f / 255.0f);
    float sd = sqrtf(fmaxf(var, 0.0f)) + 1e-6f;
    float ln = (u1 - mu) / sd * a1[j] + g1[j];
    tv[j] = fmaxf(u1 + ln, 0.0f);   // safe: block_sum256 syncs after all reads
  }
  __syncthreads();

  if (j < 64) {
    float u2 = bb2[j];
    for (int k = 0; k < 256; ++k) u2 += tv[k] * w2[k * 64 + j];
    out[(size_t)b * 64 + j] = u2;
  }
}

// --------------------------------------------------------------------------
extern "C" void kernel_launch(void* const* d_in, const int* in_sizes, int n_in,
                              void* d_out, int out_size, void* d_ws,
                              size_t ws_size, hipStream_t stream) {
  const float* x    = (const float*)d_in[0];
  const float* h0   = (const float*)d_in[1];
  const float* c0   = (const float*)d_in[2];
  const float* wih0 = (const float*)d_in[3];
  const float* whh0 = (const float*)d_in[4];
  const float* b0   = (const float*)d_in[5];
  const float* wih1 = (const float*)d_in[6];
  const float* whh1 = (const float*)d_in[7];
  const float* b1   = (const float*)d_in[8];
  const float* wfc0 = (const float*)d_in[9];
  const float* bfc0 = (const float*)d_in[10];
  const float* ln0a = (const float*)d_in[11];
  const float* ln0b = (const float*)d_in[12];
  const float* wfc1 = (const float*)d_in[13];
  const float* bfc1 = (const float*)d_in[14];
  const float* ln1a = (const float*)d_in[15];
  const float* ln1b = (const float*)d_in[16];
  const float* wfc2 = (const float*)d_in[17];
  const float* bfc2 = (const float*)d_in[18];

  char* ws = (char*)d_ws;
  _Float16* whh0s = (_Float16*)(ws + OFF_WHH0);
  _Float16* whh1s = (_Float16*)(ws + OFF_WHH1);
  _Float16* wih1s = (_Float16*)(ws + OFF_WIH1);
  _Float16* y0    = (_Float16*)(ws + OFF_Y0);
  _Float16* xp1s  = (_Float16*)(ws + OFF_XP1);
  float*    hlast = (float*)(ws + OFF_HL);

  swizzleB_kernel<<<256, 256, 0, stream>>>(whh0, whh0s, 128, 4);
  swizzleB_kernel<<<256, 256, 0, stream>>>(whh1, whh1s, 128, 4);
  swizzleB_kernel<<<512, 256, 0, stream>>>(wih1, wih1s, 256, 4);

  lstm0_kernel<<<N_ * 2, 512, LSTM_SMEM, stream>>>(x, h0, c0, wih0, b0, whh0s,
                                                   y0);
  xp1_kernel<<<N_ * 2 * T_, 512, 0, stream>>>(y0, wih1s, b1, xp1s);
  lstm1_kernel<<<N_ * 2, 512, LSTM_SMEM, stream>>>(h0, c0, whh1s, xp1s, hlast);

  head_kernel<<<B_, 256, 0, stream>>>(hlast, wfc0, bfc0, ln0a, ln0b, wfc1,
                                      bfc1, ln1a, ln1b, wfc2, bfc2,
                                      (float*)d_out);
}
